// MaskDeformConv2dModule_26079041421805
// MI455X (gfx1250) — matrix operand order repaired and checked by reading, not compiled, not run on hardware
//
#include <hip/hip_runtime.h>
#include <cstddef>

typedef __bf16 bf16;
typedef __attribute__((ext_vector_type(16))) __bf16 v16bf;
typedef __attribute__((ext_vector_type(8)))  __bf16 v8bf;
typedef __attribute__((ext_vector_type(8)))  float  v8f;

#define CIN    256
#define COUT   256
#define HW     4096
#define WID    64
#define K2     9
#define KSTEPS 72        // 9 kernel points * (256/32) channel chunks
#define MTILES 16        // 256 / 16
#define NPIX   16384     // B*H*W
#define NTILE  64        // pixels per workgroup
#define WSA_BYTES ((size_t)KSTEPS * MTILES * 32 * 16 * 2)   // 1,179,648

// ---------------------------------------------------------------------------
// Kernel 1: weights f32 [Cout][Cin][3][3] -> bf16, pre-swizzled into the exact
// v_wmma_f32_16x16x32_bf16 A-fragment layout (ISA 7.12.2, 16-bit A 16x32):
//   lane L: M = L&15 ; elem e (=2v+half): K = (v<4 ? 2v : 16+2(v-4)) + (L>=16?8:0) + half
// Stored as [kstep][mtile][lane][16 bf16] so the GEMM does contiguous b128 loads.
// ---------------------------------------------------------------------------
__global__ void swizzleW(const float* __restrict__ w, bf16* __restrict__ wsA) {
    int gid = blockIdx.x * 256 + threadIdx.x;
    if (gid >= KSTEPS * MTILES * 32 * 16) return;
    int e    = gid & 15;
    int lane = (gid >> 4) & 31;
    int mt   = (gid >> 9) & 15;
    int ks   = gid >> 13;            // 0..71
    int kp = ks >> 3, ct = ks & 7;
    int v = e >> 1, half = e & 1;
    int koff = (v < 4 ? 2 * v : 16 + 2 * (v - 4)) + ((lane >= 16) ? 8 : 0) + half;
    int c = ct * 32 + koff;          // input channel (K dim within chunk)
    int m = mt * 16 + (lane & 15);   // output channel
    wsA[gid] = (bf16)w[(m * CIN + c) * K2 + kp];
}

// ---------------------------------------------------------------------------
// Kernel 2: fused DCNv2 im2col sampling + WMMA GEMM.
// One WG = 64 output pixels x all 256 Cout. 8 waves: wm in 0..3 (64 M-rows),
// wn in 0..1 (32 pixels = 2 N-tiles each). 72 K-steps of 32 channels, with a
// double-buffered LDS B-tile (one barrier per step; next tile's gathers are
// issued before the WMMAs so their latency hides under the matrix pipe).
// ---------------------------------------------------------------------------
__launch_bounds__(256)
__global__ void dcnGemm(const float* __restrict__ x,
                        const float* __restrict__ off,
                        const float* __restrict__ msk,
                        const bf16*  __restrict__ wsA,
                        const float* __restrict__ bias,
                        float* __restrict__ convOut)
{
    __shared__ __align__(16) bf16   ldsB[2][NTILE][40]; // [buf][n][K], pad 40
    __shared__ __align__(16) int4   sIdx[K2][NTILE];
    __shared__ __align__(16) float4 sWgt[K2][NTILE];

    const int tid   = threadIdx.x;
    const int pbase = blockIdx.x * NTILE;    // 64 consecutive pixels, same batch & row
    const int b     = pbase >> 12;           // HW = 4096 = 2^12
    const int sbase = pbase & (HW - 1);

    // ---- precompute bilinear corners/weights per (kernel point, pixel) ----
    for (int t = tid; t < K2 * NTILE; t += 256) {
        int kp = t >> 6, n = t & 63;
        int s = sbase + n;
        int h = s >> 6, w = s & 63;
        float dy = off[((b * 2 * K2 + 2 * kp)     << 12) + s];
        float dx = off[((b * 2 * K2 + 2 * kp + 1) << 12) + s];
        float m  = msk[((b * K2 + kp)             << 12) + s];
        float yy = (float)(kp / 3) + (float)h - 1.0f + dy;   // STRIDE=1, PAD=1
        float xx = (float)(kp % 3) + (float)w - 1.0f + dx;
        float y0 = floorf(yy), x0 = floorf(xx);
        float fy = yy - y0,    fx = xx - x0;
        int y0i = (int)y0, x0i = (int)x0;
        int   id[4]; float wt[4];
        #pragma unroll
        for (int j = 0; j < 4; ++j) {
            int yi = y0i + (j >> 1);
            int xi = x0i + (j & 1);
            bool valid = (yi >= 0) && (yi <= WID - 1) && (xi >= 0) && (xi <= WID - 1);
            int yc = min(max(yi, 0), WID - 1), xc = min(max(xi, 0), WID - 1);
            float wy = (j >> 1) ? fy : 1.0f - fy;
            float wx = (j & 1)  ? fx : 1.0f - fx;
            id[j] = yc * WID + xc;
            wt[j] = valid ? (m * wy * wx) : 0.0f;
        }
        sIdx[kp][n] = make_int4(id[0], id[1], id[2], id[3]);
        sWgt[kp][n] = make_float4(wt[0], wt[1], wt[2], wt[3]);
    }

    const int lane = tid & 31;
    const int wid  = tid >> 5;
    const int wm   = wid & 3;                 // M group: mtiles wm*4 .. wm*4+3
    const int wn   = wid >> 2;                // N group: pixels wn*32 .. wn*32+31
    const int kb   = (lane >= 16) ? 8 : 0;    // B-fragment K base per lane half

    const int samp_n  = tid >> 2;             // 0..63 pixel
    const int samp_cq = (tid & 3) * 8;        // 8 channels per thread

    // builds the 32(K) x 64(N) bf16 tile for K-step ks into buffer bufIdx
    auto buildTile = [&](int ks, int bufIdx) {
        int kp = ks >> 3, ct = ks & 7;
        int4   id = sIdx[kp][samp_n];
        float4 wv = sWgt[kp][samp_n];
        const float* xb = x + (((size_t)b * CIN + ct * 32 + samp_cq) << 12);
        float r[8];
        #pragma unroll
        for (int j = 0; j < 8; ++j) {
            const float* p = xb + ((size_t)j << 12);
            r[j] = wv.x * p[id.x] + wv.y * p[id.y] + wv.z * p[id.z] + wv.w * p[id.w];
        }
        bf16* dst = &ldsB[bufIdx][samp_n][samp_cq];
        #pragma unroll
        for (int j = 0; j < 8; ++j) dst[j] = (bf16)r[j];
    };

    v8f acc[8];                               // [mt4][nt] -> mt4*2+nt
    #pragma unroll
    for (int i = 0; i < 8; ++i)
        #pragma unroll
        for (int j = 0; j < 8; ++j) acc[i][j] = 0.0f;

    __syncthreads();          // sIdx/sWgt ready
    buildTile(0, 0);
    __syncthreads();

    int cur = 0;
    for (int ks = 0; ks < KSTEPS; ++ks) {
        // B fragments for this wave's two N-tiles (lane holds K kb..kb+7 then kb+16..kb+23)
        v16bf bfrag[2];
        #pragma unroll
        for (int nt = 0; nt < 2; ++nt) {
            int nl = wn * 32 + nt * 16 + (lane & 15);
            v8bf blo = *(const v8bf*)&ldsB[cur][nl][kb];
            v8bf bhi = *(const v8bf*)&ldsB[cur][nl][kb + 16];
            bfrag[nt] = __builtin_shufflevector(blo, bhi,
                0, 1, 2, 3, 4, 5, 6, 7, 8, 9, 10, 11, 12, 13, 14, 15);
        }

        // kick off next tile's gathers before the WMMAs (latency overlap)
        if (ks + 1 < KSTEPS) buildTile(ks + 1, cur ^ 1);

        #pragma unroll
        for (int mt4 = 0; mt4 < 4; ++mt4) {
            v16bf afrag = *(const v16bf*)(wsA +
                ((size_t)(ks * MTILES + wm * 4 + mt4) * 32 + lane) * 16);
            acc[mt4 * 2 + 0] = __builtin_amdgcn_wmma_f32_16x16x32_bf16(
                false, afrag, false, bfrag[0], (short)0, acc[mt4 * 2 + 0], false, false);
            acc[mt4 * 2 + 1] = __builtin_amdgcn_wmma_f32_16x16x32_bf16(
                false, afrag, false, bfrag[1], (short)0, acc[mt4 * 2 + 1], false, false);
        }
        __syncthreads();      // next buffer written, current reads done
        cur ^= 1;
    }

    // ---- epilogue: C/D layout -> scatter f32 conv output (+bias) ----
    const int mhalf = (lane >= 16) ? 8 : 0;
    #pragma unroll
    for (int mt4 = 0; mt4 < 4; ++mt4) {
        #pragma unroll
        for (int nt = 0; nt < 2; ++nt) {
            int s_out = sbase + wn * 32 + nt * 16 + (lane & 15);
            int mb = wm * 64 + mt4 * 16 + mhalf;
            #pragma unroll
            for (int v = 0; v < 8; ++v) {
                int c = mb + v;               // VGPR v -> M = v (+8 upper lanes)
                convOut[(((size_t)b * COUT + c) << 12) + s_out] =
                    acc[mt4 * 2 + nt][v] + bias[c];
            }
        }
    }
}

// ---------------------------------------------------------------------------
// Kernel 3: GroupNorm stats. One WG per (b, group); the group's 8 channels x
// 4096 pixels are 32768 contiguous floats at conv + wg*32768.
// ---------------------------------------------------------------------------
__global__ void gnStats(const float* __restrict__ conv, float2* __restrict__ stats) {
    __shared__ float s1[256], s2[256];
    const float* base = conv + ((size_t)blockIdx.x << 15);
    float a = 0.0f, sq = 0.0f;
    for (int i = threadIdx.x; i < 32768; i += 256) {
        float v = base[i];
        a += v; sq += v * v;
    }
    s1[threadIdx.x] = a; s2[threadIdx.x] = sq;
    __syncthreads();
    for (int o = 128; o > 0; o >>= 1) {
        if (threadIdx.x < o) {
            s1[threadIdx.x] += s1[threadIdx.x + o];
            s2[threadIdx.x] += s2[threadIdx.x + o];
        }
        __syncthreads();
    }
    if (threadIdx.x == 0) {
        float mean = s1[0] * (1.0f / 32768.0f);
        float var  = s2[0] * (1.0f / 32768.0f) - mean * mean;
        stats[blockIdx.x] = make_float2(mean, rsqrtf(var + 1e-5f));
    }
}

// ---------------------------------------------------------------------------
// Kernel 4: normalize + affine + ReLU, in place on d_out.
// ---------------------------------------------------------------------------
__global__ void gnApply(float* __restrict__ out, const float2* __restrict__ stats,
                        const float* __restrict__ gamma, const float* __restrict__ beta) {
    int idx = blockIdx.x * 256 + threadIdx.x;            // 4,194,304 elements
    int c   = (idx >> 12) & 255;
    int bg  = ((idx >> 20) << 5) + (c >> 3);             // b*32 + group
    float2 st = stats[bg];
    float v = (out[idx] - st.x) * st.y * gamma[c] + beta[c];
    out[idx] = fmaxf(v, 0.0f);
}

// ---------------------------------------------------------------------------
extern "C" void kernel_launch(void* const* d_in, const int* in_sizes, int n_in,
                              void* d_out, int out_size, void* d_ws, size_t ws_size,
                              hipStream_t stream) {
    (void)in_sizes; (void)n_in; (void)out_size; (void)ws_size;
    const float* x      = (const float*)d_in[0];
    const float* off    = (const float*)d_in[1];
    const float* msk    = (const float*)d_in[2];
    const float* weight = (const float*)d_in[3];
    const float* bias   = (const float*)d_in[4];
    const float* gamma  = (const float*)d_in[5];
    const float* beta   = (const float*)d_in[6];
    float* out = (float*)d_out;

    bf16*   wsA   = (bf16*)d_ws;                              // 1.18 MB swizzled weights
    float2* stats = (float2*)((char*)d_ws + WSA_BYTES);       // 128 x (mean, rstd)

    swizzleW<<<(KSTEPS * MTILES * 32 * 16 + 255) / 256, 256, 0, stream>>>(weight, wsA);
    dcnGemm<<<NPIX / NTILE, 256, 0, stream>>>(x, off, msk, wsA, bias, out);
    gnStats<<<128, 256, 0, stream>>>(out, stats);
    gnApply<<<(4 * COUT * HW) / 256, 256, 0, stream>>>(out, stats, gamma, beta);
}
